// SupProtoConLoss_40235253629101
// MI455X (gfx1250) — matrix operand order repaired and checked by reading, not compiled, not run on hardware
//
#include <hip/hip_runtime.h>
#include <cstdint>

typedef __attribute__((ext_vector_type(16))) _Float16 half16;
typedef __attribute__((ext_vector_type(8)))  float    float8;

#define EPS_F      1e-8f
#define INV_TEMP   (1.0f / 0.07f)
#define THRESH_F   0.3f

constexpr int NROWS = 8192;          // B
constexpr int H     = 128;           // rep dim
constexpr int E     = 256;           // center dim
constexpr int NCLS  = 7;
constexpr int MTOT  = NROWS + NCLS;  // 8199
constexpr int MPAD  = 8208;          // 513 * 16
constexpr int RT    = NROWS / 16;    // 512 row tiles
constexpr int CT    = MPAD / 16;     // 513 col tiles
constexpr int KS    = H / 32;        // 4 k-steps of 32

// ---------------- kernel 1: curr_centers = centers @ fc_w.T + fc_b ----------
__global__ void k_centers(const float* __restrict__ centers,
                          const float* __restrict__ fc_w,
                          const float* __restrict__ fc_b,
                          float* __restrict__ cc,
                          float* __restrict__ g_acc) {
  int t = blockIdx.x * blockDim.x + threadIdx.x;
  if (t == 0) { g_acc[0] = 0.0f; g_acc[1] = 0.0f; }
  if (t >= NCLS * H) return;
  int c = t / H;
  int h = t % H;
  const float* wr = fc_w + h * E;
  const float* cr = centers + c * E;
  float s = 0.0f;
  #pragma unroll 4
  for (int e = 0; e < E; ++e) s += cr[e] * wr[e];
  cc[c * H + h] = s + fc_b[h];
}

// ---------------- kernel 2: row-normalize all_reps into rn [MPAD x 128] -----
__global__ void k_normalize(const float* __restrict__ reps,
                            const float* __restrict__ cc,
                            float* __restrict__ rn) {
  int gw   = (blockIdx.x * blockDim.x + threadIdx.x) >> 5;  // one wave per row
  int lane = threadIdx.x & 31;
  if (gw >= MPAD) return;
  float4 v = make_float4(0.f, 0.f, 0.f, 0.f);
  if (gw < NROWS)      v = *(const float4*)(reps + gw * H + lane * 4);
  else if (gw < MTOT)  v = *(const float4*)(cc + (gw - NROWS) * H + lane * 4);
  float ss = v.x * v.x + v.y * v.y + v.z * v.z + v.w * v.w;
  #pragma unroll
  for (int off = 1; off < 32; off <<= 1) ss += __shfl_xor(ss, off, 32);
  float nrm = sqrtf(ss);
  float sc  = 1.0f / fmaxf(nrm, 1e-8f);
  v.x *= sc; v.y *= sc; v.z *= sc; v.w *= sc;
  *(float4*)(rn + gw * H + lane * 4) = v;
}

// ---------------- kernel 3a: pack A fragments (hi/lo f16, ISA A-layout) -----
// A 16x32 f16 layout: lane l holds row (l&15); slot s -> k = (s&7) + 8*(((s>>3)<<1)|(l>>4))
__global__ void k_packA(const float* __restrict__ rn,
                        _Float16* __restrict__ ah, _Float16* __restrict__ al) {
  int t = blockIdx.x * blockDim.x + threadIdx.x;
  if (t >= RT * KS * 32 * 16) return;
  int slot = t & 15;
  int lane = (t >> 4) & 31;
  int ks   = (t >> 9) & 3;
  int tile = t >> 11;
  int row  = tile * 16 + (lane & 15);
  int kl   = (slot & 7) + 8 * (((slot >> 3) << 1) | (lane >> 4));
  float x  = rn[row * H + ks * 32 + kl];
  _Float16 h = (_Float16)x;
  ah[t] = h;
  al[t] = (_Float16)(x - (float)h);
}

// ---------------- kernel 3b: pack B fragments (hi/lo f16, ISA B-layout) -----
// B 32x16 f16 layout: lane l holds col (l&15); slot s -> k = (s&7) + 8*(((s>>3)<<1)|(l>>4))
__global__ void k_packB(const float* __restrict__ rn,
                        _Float16* __restrict__ bh, _Float16* __restrict__ bl) {
  int t = blockIdx.x * blockDim.x + threadIdx.x;
  if (t >= CT * KS * 32 * 16) return;
  int slot = t & 15;
  int lane = (t >> 4) & 31;
  int ks   = (t >> 9) & 3;
  int tile = t >> 11;
  int col  = tile * 16 + (lane & 15);
  int k    = ks * 32 + (slot & 7) + 8 * (((slot >> 3) << 1) | (lane >> 4));
  float x  = rn[col * H + k];
  _Float16 h = (_Float16)x;
  bh[t] = h;
  bl[t] = (_Float16)(x - (float)h);
}

// ---------------- kernel 4: WMMA GEMM + fused masked softmax-ratio loss -----
__global__ __launch_bounds__(256) void k_gemm_loss(
    const _Float16* __restrict__ ah, const _Float16* __restrict__ al,
    const _Float16* __restrict__ bh, const _Float16* __restrict__ bl,
    const int* __restrict__ labels, float* __restrict__ g_acc) {
  int tile = blockIdx.x;            // row tile (16 rows)
  int wave = threadIdx.x >> 5;      // 8 waves split the column tiles
  int lane = threadIdx.x & 31;
  int hi8  = (lane >> 4) << 3;      // lanes 16-31 hold rows +8
  int rowbase = tile * 16 + hi8;

  // resident A fragments (hi + lo) for all 4 k-steps
  half16 Ah[KS], Al[KS];
  #pragma unroll
  for (int ks = 0; ks < KS; ++ks) {
    int off = ((tile * KS + ks) * 32 + lane) * 16;
    Ah[ks] = *(const half16*)(ah + off);
    Al[ks] = *(const half16*)(al + off);
  }
  int labRow[8];
  #pragma unroll
  for (int v = 0; v < 8; ++v) labRow[v] = labels[rowbase + v];

  float accp[8], accn[8], accc[8];
  #pragma unroll
  for (int v = 0; v < 8; ++v) { accp[v] = 0.f; accn[v] = 0.f; accc[v] = 0.f; }

  for (int ct = wave; ct < CT; ct += 8) {
    float8 c0 = {}, c1 = {}, c2 = {};
    #pragma unroll
    for (int ks = 0; ks < KS; ++ks) {
      int off = ((ct * KS + ks) * 32 + lane) * 16;
      half16 Bh = *(const half16*)(bh + off);
      half16 Bl = *(const half16*)(bl + off);
      // split-f16 error-corrected product: Ah*Bh + Ah*Bl + Al*Bh (3 indep chains)
      c0 = __builtin_amdgcn_wmma_f32_16x16x32_f16(false, Ah[ks], false, Bh,
                                                  (short)0, c0, false, false);
      c1 = __builtin_amdgcn_wmma_f32_16x16x32_f16(false, Ah[ks], false, Bl,
                                                  (short)0, c1, false, false);
      c2 = __builtin_amdgcn_wmma_f32_16x16x32_f16(false, Al[ks], false, Bh,
                                                  (short)0, c2, false, false);
    }
    int col = ct * 16 + (lane & 15);
    if (col < MTOT) {
      int labc = (col < NROWS) ? labels[col] : (col - NROWS);
      #pragma unroll
      for (int v = 0; v < 8; ++v) {
        int row = rowbase + v;
        float cosv = c0[v] + c1[v] + c2[v];
        // score - 1/TEMP (uniform shift; ratio is shift-invariant, exp <= ~1)
        float s = (0.5f * cosv - 0.5f + EPS_F) * INV_TEMP;
        float e = __expf(s);
        bool pos = (labRow[v] == labc);
        if (row != col) {  // diagonal never contributes (masked from both sums)
          if (pos) { accp[v] += e; accc[v] += 1.0f; }
          else     { accn[v] += e; }
        }
      }
    }
  }

  // butterfly reduce within 16-lane halves (offsets < 16 keep halves separate)
  #pragma unroll
  for (int v = 0; v < 8; ++v) {
    #pragma unroll
    for (int off = 1; off < 16; off <<= 1) {
      accp[v] += __shfl_xor(accp[v], off, 32);
      accn[v] += __shfl_xor(accn[v], off, 32);
      accc[v] += __shfl_xor(accc[v], off, 32);
    }
  }

  __shared__ float sp[16], sn[16], sc[16];
  __shared__ float s_out[2];
  if (threadIdx.x < 16) { sp[threadIdx.x] = 0.f; sn[threadIdx.x] = 0.f; sc[threadIdx.x] = 0.f; }
  if (threadIdx.x == 0) { s_out[0] = 0.f; s_out[1] = 0.f; }
  __syncthreads();
  #pragma unroll
  for (int v = 0; v < 8; ++v) {
    if ((lane & 15) == v) {       // lanes v and v+16 own rows v and v+8
      int r = v + hi8;
      atomicAdd(&sp[r], accp[v]);
      atomicAdd(&sn[r], accn[v]);
      atomicAdd(&sc[r], accc[v]);
    }
  }
  __syncthreads();
  if (threadIdx.x < 16) {
    float p = sp[threadIdx.x], n = sn[threadIdx.x], cnt = sc[threadIdx.x];
    float probs = p / (p + n);
    probs = probs / (cnt + EPS_F);
    float lv = -logf(probs + EPS_F);
    if (lv > THRESH_F) { atomicAdd(&s_out[0], lv); atomicAdd(&s_out[1], 1.0f); }
  }
  __syncthreads();
  if (threadIdx.x == 0) {
    atomicAdd(&g_acc[0], s_out[0]);
    atomicAdd(&g_acc[1], s_out[1]);
  }
}

// ---------------- kernel 5: final scalar -----------------------------------
__global__ void k_final(const float* __restrict__ g_acc, float* __restrict__ out) {
  out[0] = g_acc[0] / (g_acc[1] + EPS_F);
}

// ---------------- launcher --------------------------------------------------
extern "C" void kernel_launch(void* const* d_in, const int* in_sizes, int n_in,
                              void* d_out, int out_size, void* d_ws, size_t ws_size,
                              hipStream_t stream) {
  const float* reps    = (const float*)d_in[0];  // [8192,128]
  const int*   labels  = (const int*)d_in[1];    // [8192]
  const float* centers = (const float*)d_in[2];  // [7,256]
  const float* fc_w    = (const float*)d_in[3];  // [128,256]
  const float* fc_b    = (const float*)d_in[4];  // [128]
  float* out = (float*)d_out;

  // workspace layout (all offsets 256B-aligned)
  char* w = (char*)d_ws;
  constexpr size_t OFF_CC  = 0;                         // 7*128*4   = 3584 -> 4096
  constexpr size_t OFF_RN  = 4096;                      // 8208*128*4 = 4202496
  constexpr size_t OFF_AH  = OFF_RN + 4202496;          // 512*4*32*16*2 = 2097152
  constexpr size_t OFF_AL  = OFF_AH + 2097152;
  constexpr size_t OFF_BH  = OFF_AL + 2097152;          // 513*4*32*16*2 = 2101248
  constexpr size_t OFF_BL  = OFF_BH + 2101248;
  constexpr size_t OFF_ACC = OFF_BL + 2101248;

  float*    cc   = (float*)(w + OFF_CC);
  float*    rn   = (float*)(w + OFF_RN);
  _Float16* ahp  = (_Float16*)(w + OFF_AH);
  _Float16* alp  = (_Float16*)(w + OFF_AL);
  _Float16* bhp  = (_Float16*)(w + OFF_BH);
  _Float16* blp  = (_Float16*)(w + OFF_BL);
  float*    gacc = (float*)(w + OFF_ACC);

  k_centers<<<(NCLS * H + 255) / 256, 256, 0, stream>>>(centers, fc_w, fc_b, cc, gacc);
  k_normalize<<<(MPAD * 32) / 256, 256, 0, stream>>>(reps, cc, rn);
  k_packA<<<(RT * KS * 32 * 16) / 256, 256, 0, stream>>>(rn, ahp, alp);
  k_packB<<<(CT * KS * 32 * 16) / 256, 256, 0, stream>>>(rn, bhp, blp);
  k_gemm_loss<<<RT, 256, 0, stream>>>(ahp, alp, bhp, blp, labels, gacc);
  k_final<<<1, 1, 0, stream>>>(gacc, out);
}
